// EgoNet_31018253812029
// MI455X (gfx1250) — matrix operand order repaired and checked by reading, not compiled, not run on hardware
//
#include <hip/hip_runtime.h>
#include <hip/hip_bf16.h>
#include <math.h>

#define B_    4
#define C_    32
#define Z_    64
#define Y_    32
#define X_    64
#define R_    3
#define SPAT  (Z_*Y_*X_)        // 131072 per (n, c)
#define NV    (B_*R_)           // 12 warped volumes
#define TILES 128               // spatial tiles per n for partial sums
#define POS_PER_BLOCK (SPAT/TILES)  // 1024
#define EPSF  1e-6f
#define INV_DEN (1.0f/(32.0f + 1e-6f))
#define PI_F  3.14159265358979323846f

typedef __attribute__((ext_vector_type(16))) _Float16 v16h;
typedef __attribute__((ext_vector_type(8)))  float    v8f;

// ---------------------------------------------------------------------------
// Kernel 1: fused rotate-warp (bilinear in x,z; y is exact) + 27-shift
// correlation, reduced to per-block partial sums. Lanes cover consecutive x
// -> all global loads coalesced; both volumes live in the 192MB L2.
// ---------------------------------------------------------------------------
__global__ __launch_bounds__(256) void heat_partial_kernel(
    const float* __restrict__ feat0, const float* __restrict__ feat1,
    float* __restrict__ partial)
{
  const int tile = blockIdx.x;
  const int n    = blockIdx.y;         // n = b*R_ + r
  const int b    = n / R_;
  const int r    = n % R_;
  const float deg = -4.0f + 4.0f * (float)r;   // linspace(-4,4,3)
  const float th  = deg * (PI_F / 180.0f);
  const float cs  = cosf(th);
  const float sn  = sinf(th);
  const float cx = (X_-1)*0.5f, cz = (Z_-1)*0.5f;

  const float* f1b = feat1 + (long)b * C_ * SPAT;
  const float* f0b = feat0 + (long)b * C_ * SPAT;

  float acc[27];
  #pragma unroll
  for (int i = 0; i < 27; ++i) acc[i] = 0.0f;

  for (int it = 0; it < POS_PER_BLOCK/256; ++it) {
    const int idx = tile * POS_PER_BLOCK + it*256 + threadIdx.x;
    const int x = idx & (X_-1);
    const int y = (idx >> 6) & (Y_-1);
    const int z = idx >> 11;

    // src = R^T * (p - center) + center;  R = rot_y -> y unchanged
    const float rx = (float)x - cx, rz = (float)z - cz;
    const float sx = cs*rx - sn*rz + cx;
    const float sz = sn*rx + cs*rz + cz;
    const float fx0 = floorf(sx), fz0 = floorf(sz);
    const int ix0 = (int)fx0, iz0 = (int)fz0;
    const float fx = sx - fx0, fz = sz - fz0;

    int   toff[4]; float twgt[4];
    #pragma unroll
    for (int t = 0; t < 4; ++t) {
      const int dxt = t & 1, dzt = t >> 1;
      const int ix = ix0 + dxt, iz = iz0 + dzt;
      const float w = (dxt ? fx : 1.0f-fx) * (dzt ? fz : 1.0f-fz);
      const bool v = (ix >= 0) && (ix < X_) && (iz >= 0) && (iz < Z_);
      const int ixc = min(max(ix, 0), X_-1);
      const int izc = min(max(iz, 0), Z_-1);
      toff[t] = (izc*Y_ + y)*X_ + ixc;
      twgt[t] = v ? w : 0.0f;
    }

    // feat0 neighbor offsets for the 27 shifts (p = q - off, reference order
    // si = dz*9 + dy*3 + dx with offsets in {-1,0,1})
    int poff[27]; float pmsk[27];
    #pragma unroll
    for (int s = 0; s < 27; ++s) {
      const int dz = s/9 - 1, dy = (s/3)%3 - 1, dx = s%3 - 1;
      const int pz = z - dz, py = y - dy, px = x - dx;
      const bool v = (pz >= 0) && (pz < Z_) && (py >= 0) && (py < Y_)
                  && (px >= 0) && (px < X_);
      poff[s] = v ? ((pz*Y_ + py)*X_ + px) : 0;
      pmsk[s] = v ? 1.0f : 0.0f;
    }

    for (int ch = 0; ch < C_; ++ch) {
      const float* __restrict__ f1c = f1b + (long)ch * SPAT;
      const float* __restrict__ f0c = f0b + (long)ch * SPAT;
      const float fN = twgt[0]*f1c[toff[0]] + twgt[1]*f1c[toff[1]]
                     + twgt[2]*f1c[toff[2]] + twgt[3]*f1c[toff[3]];
      #pragma unroll
      for (int s = 0; s < 27; ++s) {
        acc[s] = fmaf(pmsk[s] * f0c[poff[s]], fN, acc[s]);
      }
    }
  }

  // fixed-order tree reduction (deterministic, no float atomics)
  __shared__ float red[256];
  for (int s = 0; s < 27; ++s) {
    red[threadIdx.x] = acc[s];
    __syncthreads();
    for (int off = 128; off > 0; off >>= 1) {
      if (threadIdx.x < off) red[threadIdx.x] += red[threadIdx.x + off];
      __syncthreads();
    }
    if (threadIdx.x == 0)
      partial[((long)n*TILES + tile)*27 + s] = red[0];
    __syncthreads();
  }
}

// ---------------------------------------------------------------------------
// WMMA operand marshalling per CDNA5 ISA 7.12.2 (wave32), reading f16 LDS
// staging buffers (zero-padded) -> unconditional ds_loads, no exec juggling.
// A 16x32 f16: lane L -> M=L&15; element e -> K = e + 16*(L>=16) (contiguous order used here, not the ISA table order).
// B 32x16 f16: col N = L&15; element e -> K = e + 16*(L>=16) (same contiguous order as A).
// C/D f32 16x16: col N = L&15; vgpr r -> M = r + 8*(L>=16).
// ---------------------------------------------------------------------------
__device__ inline v16h build_a(const _Float16* act, int kpad, int lane, int k0) {
  v16h a;
  const int  m  = lane & 15;
  const bool hi = lane >= 16;
  #pragma unroll
  for (int e = 0; e < 16; ++e) {
    const int k = e + (hi ? 16 : 0);
    a[e] = act[m * kpad + k0 + k];
  }
  return a;
}

__device__ inline v16h build_b(const _Float16* Wh, int kpad, int jbase,
                               int lane, int k0) {
  v16h bv;
  const int j    = jbase + (lane & 15);
  const int kadd = (lane < 16) ? 0 : 16;
  #pragma unroll
  for (int e = 0; e < 16; ++e) {
    bv[e] = Wh[j * kpad + k0 + e + kadd];
  }
  return bv;
}

// ---------------------------------------------------------------------------
// Kernel 2: single-wave. Reduce partials -> heat -> f (lrelu + L2 norm in
// f32), then 3-layer MLP via v_wmma_f32_16x16x32_f16 with all operands staged
// in zero-padded f16 LDS (M padded 4->16, K padded to 32-multiples), then
// build T_e and the scalar loss.
// ---------------------------------------------------------------------------
__global__ __launch_bounds__(32) void mlp_kernel(
    const float* __restrict__ partial, const float* __restrict__ g,
    const float* __restrict__ W1, const float* __restrict__ b1,
    const float* __restrict__ W2, const float* __restrict__ b2,
    const float* __restrict__ W3, const float* __restrict__ b3,
    float* __restrict__ out)
{
  __shared__ _Float16 W1h[128*96];    // W1 padded 81 -> 96 K
  __shared__ _Float16 W2h[128*128];
  __shared__ _Float16 W3h[16*128];    // W3 padded 4 -> 16 N rows
  __shared__ _Float16 actA[16*96];
  __shared__ _Float16 act1[16*128];
  __shared__ _Float16 act2[16*128];
  __shared__ float    fbuf[4*81];     // f32 staging for normalization
  __shared__ float    invn[4];
  __shared__ float    oby[16];
  const int lane = threadIdx.x;

  // ---- stage weights (zero-padded) into f16 LDS ----
  for (int i = lane; i < 128*96; i += 32) {
    const int j = i / 96, k = i % 96;
    W1h[i] = (k < 81) ? (_Float16)W1[j*81 + k] : (_Float16)0.0f;
  }
  for (int i = lane; i < 128*128; i += 32) W2h[i] = (_Float16)W2[i];
  for (int i = lane; i < 16*128; i += 32) {
    const int j = i / 128, k = i % 128;
    W3h[i] = (j < 4) ? (_Float16)W3[j*128 + k] : (_Float16)0.0f;
  }
  for (int i = lane; i < 16*96;  i += 32) actA[i] = (_Float16)0.0f;
  for (int i = lane; i < 16*128; i += 32) {
    act1[i] = (_Float16)0.0f;
    act2[i] = (_Float16)0.0f;
  }
  __syncthreads();

  // ---- heat reduce (fixed order) + scale + leaky_relu -> fbuf (f32) ----
  for (int id = lane; id < NV*27; id += 32) {
    const int n = id / 27, s = id % 27;
    float v = 0.0f;
    for (int t = 0; t < TILES; ++t) v += partial[((long)n*TILES + t)*27 + s];
    v *= INV_DEN;
    v = (v > 0.0f) ? v : 0.1f * v;
    const int b = n / R_, r = n % R_;
    fbuf[b*81 + r*27 + s] = v;         // f[b, r*27+s]
  }
  __syncthreads();
  if (lane < 4) {
    float ss = 0.0f;
    for (int k = 0; k < 81; ++k) { const float v = fbuf[lane*81 + k]; ss += v*v; }
    invn[lane] = 1.0f / (EPSF + sqrtf(ss));
  }
  __syncthreads();
  for (int id = lane; id < 4*81; id += 32) {
    const int b = id / 81, k = id % 81;
    actA[b*96 + k] = (_Float16)(fbuf[id] * invn[b]);
  }
  __syncthreads();

  const int nn = lane & 15;

  // ---- Layer 1: (16x96) @ W1^T -> 128, lrelu ----
  for (int jb = 0; jb < 8; ++jb) {
    v8f c = {};
    for (int kb = 0; kb < 3; ++kb) {
      const v16h a  = build_a(actA, 96, lane, kb*32);
      const v16h bm = build_b(W1h, 96, jb*16, lane, kb*32);
      c = __builtin_amdgcn_wmma_f32_16x16x32_f16(false, a, false, bm,
                                                 (short)0, c, false, false);
    }
    if (lane < 16) {
      #pragma unroll
      for (int rr = 0; rr < 4; ++rr) {
        const float v = c[rr] + b1[jb*16 + nn];
        act1[rr*128 + jb*16 + nn] = (_Float16)((v > 0.0f) ? v : 0.1f*v);
      }
    }
    __syncthreads();
  }

  // ---- Layer 2: (16x128) @ W2^T -> 128, lrelu ----
  for (int jb = 0; jb < 8; ++jb) {
    v8f c = {};
    for (int kb = 0; kb < 4; ++kb) {
      const v16h a  = build_a(act1, 128, lane, kb*32);
      const v16h bm = build_b(W2h, 128, jb*16, lane, kb*32);
      c = __builtin_amdgcn_wmma_f32_16x16x32_f16(false, a, false, bm,
                                                 (short)0, c, false, false);
    }
    if (lane < 16) {
      #pragma unroll
      for (int rr = 0; rr < 4; ++rr) {
        const float v = c[rr] + b2[jb*16 + nn];
        act2[rr*128 + jb*16 + nn] = (_Float16)((v > 0.0f) ? v : 0.1f*v);
      }
    }
    __syncthreads();
  }

  // ---- Layer 3: (16x128) @ W3^T -> 4 (N padded to 16 with zero rows) ----
  {
    v8f c = {};
    for (int kb = 0; kb < 4; ++kb) {
      const v16h a  = build_a(act2, 128, lane, kb*32);
      const v16h bm = build_b(W3h, 128, 0, lane, kb*32);
      c = __builtin_amdgcn_wmma_f32_16x16x32_f16(false, a, false, bm,
                                                 (short)0, c, false, false);
    }
    if (lane < 4) {
      #pragma unroll
      for (int rr = 0; rr < 4; ++rr) oby[rr*4 + lane] = c[rr] + b3[lane];
    }
    __syncthreads();
  }

  // ---- T_e + loss (lane 0, fixed order) ----
  if (lane == 0) {
    float loss = 0.0f;
    for (int b = 0; b < 4; ++b) {
      const float rdeg = oby[b*4+0];
      const float yv   = oby[b*4+1];
      const float xv   = oby[b*4+2];
      const float zv   = oby[b*4+3];
      const float th = rdeg * (PI_F/180.0f);
      const float cc = cosf(th), sn = sinf(th);
      float* T = out + 1 + b*16;     // cam0_T_cam1_e, row-major 4x4
      T[0]=cc;   T[1]=0.0f; T[2]=sn;   T[3]=-xv;
      T[4]=0.0f; T[5]=1.0f; T[6]=0.0f; T[7]=-yv;
      T[8]=-sn;  T[9]=0.0f; T[10]=cc;  T[11]=-zv;
      T[12]=0.0f;T[13]=0.0f;T[14]=0.0f;T[15]=1.0f;
      const float deg_e = atan2f(sn, cc) * (180.0f/PI_F);
      const float* G = g + b*16;
      const float deg_g = atan2f(G[2], G[10]) * (180.0f/PI_F);
      const float dx = -xv - G[3], dy = -yv - G[7], dz = -zv - G[11];
      const float dd = deg_e - deg_g;
      loss += 0.25f * (dx*dx + dy*dy + dz*dz) + 0.25f * dd*dd;
    }
    out[0] = loss;
  }
}

// ---------------------------------------------------------------------------
// Kernel 3: final trilinear warp of feat1 by T_e (read from d_out[1..64]).
// One thread per spatial position, loops 32 channels; stores coalesced.
// ---------------------------------------------------------------------------
__global__ __launch_bounds__(256) void warp_kernel(
    const float* __restrict__ feat1, const float* __restrict__ out_T,
    float* __restrict__ warped)
{
  const int b   = blockIdx.y;
  const int idx = blockIdx.x * 256 + threadIdx.x;
  const float* T = out_T + 1 + b*16;
  const float R00=T[0], R01=T[1], R02=T[2],  tx=T[3];
  const float R10=T[4], R11=T[5], R12=T[6],  ty=T[7];
  const float R20=T[8], R21=T[9], R22=T[10], tz=T[11];
  // invert_rt: rinv = R^T, tinv = -R^T t
  const float I00=R00, I01=R10, I02=R20;
  const float I10=R01, I11=R11, I12=R21;
  const float I20=R02, I21=R12, I22=R22;
  const float tix = -(I00*tx + I01*ty + I02*tz);
  const float tiy = -(I10*tx + I11*ty + I12*tz);
  const float tiz = -(I20*tx + I21*ty + I22*tz);

  const int x = idx & 63, y = (idx >> 6) & 31, z = idx >> 11;
  const float cxc = 31.5f, cyc = 15.5f, czc = 31.5f;
  const float rx = (float)x - cxc, ry = (float)y - cyc, rz = (float)z - czc;
  const float sx = I00*rx + I01*ry + I02*rz + tix + cxc;
  const float sy = I10*rx + I11*ry + I12*rz + tiy + cyc;
  const float sz = I20*rx + I21*ry + I22*rz + tiz + czc;
  const float fx0 = floorf(sx), fy0 = floorf(sy), fz0 = floorf(sz);
  const int ix0 = (int)fx0, iy0 = (int)fy0, iz0 = (int)fz0;
  const float fx = sx - fx0, fy = sy - fy0, fz = sz - fz0;

  int toff[8]; float twgt[8];
  #pragma unroll
  for (int t = 0; t < 8; ++t) {
    const int dxt = t & 1, dyt = (t >> 1) & 1, dzt = t >> 2;
    const int ix = ix0 + dxt, iy = iy0 + dyt, iz = iz0 + dzt;
    const float w = (dxt ? fx : 1.0f-fx) * (dyt ? fy : 1.0f-fy)
                  * (dzt ? fz : 1.0f-fz);
    const bool v = (ix >= 0) && (ix < 64) && (iy >= 0) && (iy < 32)
                && (iz >= 0) && (iz < 64);
    const int ixc = min(max(ix, 0), 63);
    const int iyc = min(max(iy, 0), 31);
    const int izc = min(max(iz, 0), 63);
    toff[t] = (izc*32 + iyc)*64 + ixc;
    twgt[t] = v ? w : 0.0f;
  }

  const float* f1b = feat1  + (long)b * C_ * SPAT;
  float*       ob  = warped + (long)b * C_ * SPAT;
  for (int ch = 0; ch < C_; ++ch) {
    const float* __restrict__ f1c = f1b + (long)ch * SPAT;
    float v = 0.0f;
    #pragma unroll
    for (int t = 0; t < 8; ++t) v = fmaf(twgt[t], f1c[toff[t]], v);
    ob[(long)ch * SPAT + idx] = v;
  }
}

// ---------------------------------------------------------------------------
extern "C" void kernel_launch(void* const* d_in, const int* in_sizes, int n_in,
                              void* d_out, int out_size, void* d_ws, size_t ws_size,
                              hipStream_t stream) {
  (void)in_sizes; (void)n_in; (void)out_size; (void)ws_size;
  const float* feat0 = (const float*)d_in[0];
  const float* feat1 = (const float*)d_in[1];
  const float* g     = (const float*)d_in[2];
  const float* W1    = (const float*)d_in[3];
  const float* b1    = (const float*)d_in[4];
  const float* W2    = (const float*)d_in[5];
  const float* b2    = (const float*)d_in[6];
  const float* W3    = (const float*)d_in[7];
  const float* b3    = (const float*)d_in[8];
  float* out     = (float*)d_out;            // [0]=loss, [1..64]=T_e, [65..]=warped
  float* partial = (float*)d_ws;             // 12*128*27 floats = 166 KB

  heat_partial_kernel<<<dim3(TILES, NV), 256, 0, stream>>>(feat0, feat1, partial);
  mlp_kernel<<<1, 32, 0, stream>>>(partial, g, W1, b1, W2, b2, W3, b3, out);
  warp_kernel<<<dim3(SPAT/256, B_), 256, 0, stream>>>(feat1, out, out + 65);
}
